// MultiQueryAttention_13082470384178
// MI455X (gfx1250) — matrix operand order repaired and checked by reading, not compiled, not run on hardware
//
#include <hip/hip_runtime.h>
#include <stdint.h>

#define DEV static __device__ __forceinline__

typedef __attribute__((ext_vector_type(16))) __bf16 v16bf;
typedef __attribute__((ext_vector_type(8)))  float  v8f;

union Frag {
  v16bf bf;
  unsigned int u[8];
};

DEV unsigned short f2bf(float x) {
  unsigned int u = __float_as_uint(x);
  unsigned int r = u + 0x7fffu + ((u >> 16) & 1u);   // round-to-nearest-even
  return (unsigned short)(r >> 16);
}
DEV unsigned int pkbf(float lo, float hi) {
  return (unsigned int)f2bf(lo) | ((unsigned int)f2bf(hi) << 16);
}
DEV v8f vzero() {
  v8f z;
#pragma unroll
  for (int r = 0; r < 8; ++r) z[r] = 0.0f;
  return z;
}

// A-fragment (16x32 bf16, row-major source): lane holds row `row`,
// 16 contiguous K at k0 + hi*16.
DEV Frag load_a_frag(const unsigned short* __restrict__ base, int ld, int row, int k0, int hi) {
  Frag f;
  const unsigned short* p = base + (size_t)row * ld + k0 + hi * 16;
  uint4 a = *(const uint4*)p;
  uint4 b = *(const uint4*)(p + 8);
  f.u[0] = a.x; f.u[1] = a.y; f.u[2] = a.z; f.u[3] = a.w;
  f.u[4] = b.x; f.u[5] = b.y; f.u[6] = b.z; f.u[7] = b.w;
  return f;
}
// B-fragment (32x16 bf16) where column n = row `row` of a row-major matrix:
// lane holds 16 contiguous bf16 at k0 + hi*16.
DEV Frag load_b_frag(const unsigned short* __restrict__ base, int ld, int row, int k0, int hi) {
  Frag f;
  const unsigned short* p = base + (size_t)row * ld + k0 + hi * 16;
  uint4 a = *(const uint4*)p;
  uint4 b = *(const uint4*)(p + 8);
  f.u[0] = a.x; f.u[1] = a.y; f.u[2] = a.z; f.u[3] = a.w;
  f.u[4] = b.x; f.u[5] = b.y; f.u[6] = b.z; f.u[7] = b.w;
  return f;
}
DEV v8f wmma_bf16(const Frag& a, const Frag& b, v8f c) {
  return __builtin_amdgcn_wmma_f32_16x16x32_bf16(false, a.bf, false, b.bf,
                                                 (short)0, c, false, false);
}

// ---- CDNA5 async memory->LDS DMA (ASYNCcnt path), inline asm ----
DEV void async_b128(unsigned lds_byte_off, const unsigned short* g) {
  unsigned long long ga = (unsigned long long)(uintptr_t)g;
  asm volatile("global_load_async_to_lds_b128 %0, %1, off"
               :: "v"(lds_byte_off), "v"(ga) : "memory");
}
DEV void wait_async0() { asm volatile("s_wait_asynccnt 0x0" ::: "memory"); }
DEV void wait_ds0()    { asm volatile("s_wait_dscnt 0x0" ::: "memory"); }

// ---------------- fp32 -> bf16 convert ----------------
__global__ void k_cvt_bf16(const float* __restrict__ in, unsigned short* __restrict__ out, int n) {
  int i = blockIdx.x * blockDim.x + threadIdx.x;
  if (i < n) out[i] = f2bf(in[i]);
}

// ---------------- C = A(MxK) @ W(NxK)^T, bf16 in, f32 out ----------------
// Block 256 thr = 8 waves (4 M x 2 N); wave tile 32x64; block tile 128x128.
// Double-buffered LDS staging (k-step 64) via global_load_async_to_lds_b128:
// 16 WMMAs per barrier, async DMA of the next 36 KB tile overlapped with compute.
#define SM_LDU   72                 // padded row stride in ushorts (144 B, 16B-aligned)
#define SM_BYTES (128 * SM_LDU * 2) // bytes per buffer (18 KB)

__global__ __launch_bounds__(256) void k_gemm_bf16_nt(
    const unsigned short* __restrict__ A, const unsigned short* __restrict__ W,
    float* __restrict__ C, int M, int N, int K) {
  __shared__ unsigned short smA[2][128 * SM_LDU];
  __shared__ unsigned short smB[2][128 * SM_LDU];

  const int tid  = threadIdx.x;
  const int lane = tid & 31;
  const int wave = tid >> 5;
  const int hi = lane >> 4;
  const int lr = lane & 15;
  const int wm = wave >> 1;
  const int wn = wave & 1;
  const int mblk = blockIdx.y * 128;
  const int nblk = blockIdx.x * 128;

  // cooperative tile loader: thread copies 64 B of A and 64 B of B per k-step
  const int ldrow = tid >> 1;        // 0..127
  const int ldcol = (tid & 1) * 32;  // ushort offset inside 64-wide tile
  const unsigned short* gA = A + (size_t)(mblk + ldrow) * K + ldcol;
  const unsigned short* gB = W + (size_t)(nblk + ldrow) * K + ldcol;
  const unsigned ldsA = (unsigned)(uintptr_t)&smA[0][ldrow * SM_LDU + ldcol];
  const unsigned ldsB = (unsigned)(uintptr_t)&smB[0][ldrow * SM_LDU + ldcol];

  v8f acc[2][4];
#pragma unroll
  for (int i = 0; i < 2; ++i)
#pragma unroll
    for (int j = 0; j < 4; ++j) acc[i][j] = vzero();

  // prologue: stage tile for k0 = 0 into buffer 0
#pragma unroll
  for (int c = 0; c < 4; ++c) {
    async_b128(ldsA + 16 * c, gA + 8 * c);
    async_b128(ldsB + 16 * c, gB + 8 * c);
  }

  int buf = 0;
  for (int k0 = 0; k0 < K; k0 += 64) {
    wait_async0();   // my portion of tile `buf` landed in LDS
    wait_ds0();      // my reads of buffer buf^1 (prev iter) fully drained
    __syncthreads(); // -> whole tile valid, other buffer reusable, block-wide

    if (k0 + 64 < K) {
      const unsigned dst = (buf ^ 1) ? SM_BYTES : 0u;
      const unsigned short* a = gA + k0 + 64;
      const unsigned short* b = gB + k0 + 64;
#pragma unroll
      for (int c = 0; c < 4; ++c) {
        async_b128(ldsA + dst + 16 * c, a + 8 * c);
        async_b128(ldsB + dst + 16 * c, b + 8 * c);
      }
    }

#pragma unroll
    for (int kk = 0; kk < 64; kk += 32) {
      Frag af[2], bf[4];
#pragma unroll
      for (int i = 0; i < 2; ++i)
        af[i] = load_a_frag(smA[buf], SM_LDU, wm * 32 + 16 * i + lr, kk, hi);
#pragma unroll
      for (int j = 0; j < 4; ++j)
        bf[j] = load_b_frag(smB[buf], SM_LDU, wn * 64 + 16 * j + lr, kk, hi);
#pragma unroll
      for (int i = 0; i < 2; ++i)
#pragma unroll
        for (int j = 0; j < 4; ++j) acc[i][j] = wmma_bf16(af[i], bf[j], acc[i][j]);
    }

    buf ^= 1;
  }

  // C layout: lane l, VGPR r -> (m0+16i + r + 8*hi, n0+16j + lr)
#pragma unroll
  for (int i = 0; i < 2; ++i)
#pragma unroll
    for (int j = 0; j < 4; ++j) {
      float* cp = C + (size_t)(mblk + wm * 32 + 16 * i + 8 * hi) * N + nblk + wn * 64 + 16 * j + lr;
#pragma unroll
      for (int r = 0; r < 8; ++r) cp[(size_t)r * N] = acc[i][j][r];
    }
}

// ---------------- RoPE + RMS-norm on q, emit bf16 (B,H,L,128) ----------------
__global__ void k_rope_rms_q(const float* __restrict__ qf, const float* __restrict__ cs,
                             const float* __restrict__ sn, unsigned short* __restrict__ qb,
                             int B, int L, int H) {
  int task = blockIdx.x * 8 + (threadIdx.x >> 5);     // task = (b*L + l)*H + h
  int lane = threadIdx.x & 31;
  if (task >= B * L * H) return;
  int h = task % H;
  int bl = task / H;
  int l = bl % L;
  int b = bl / L;
  const float* row = qf + (size_t)bl * 2048 + h * 128;
  float x0 = row[lane], x1 = row[lane + 32], x2 = row[lane + 64], x3 = row[lane + 96];
  float c0 = cs[(size_t)l * 64 + lane], c1 = cs[(size_t)l * 64 + lane + 32];
  float s0 = sn[(size_t)l * 64 + lane], s1 = sn[(size_t)l * 64 + lane + 32];
  float y0 = x0 * c0 + x2 * s0;
  float y1 = x1 * c1 + x3 * s1;
  float y2 = -x0 * s0 + x2 * c0;
  float y3 = -x1 * s1 + x3 * c1;
  float ss = y0 * y0 + y1 * y1 + y2 * y2 + y3 * y3;
#pragma unroll
  for (int off = 16; off; off >>= 1) ss += __shfl_xor(ss, off, 32);
  float r = rsqrtf(ss * (1.0f / 128.0f) + 1.1920929e-07f);
  unsigned short* o = qb + (((size_t)b * H + h) * L + l) * 128;
  o[lane] = f2bf(y0 * r); o[lane + 32] = f2bf(y1 * r);
  o[lane + 64] = f2bf(y2 * r); o[lane + 96] = f2bf(y3 * r);
}

// ---- RoPE+RMS on k (-> new_k f32 + bf16), pass v (-> new_v f32 + bf16 V^T) ----
__global__ void k_rope_rms_kv(const float* __restrict__ kf, const float* __restrict__ vf,
                              const float* __restrict__ cs, const float* __restrict__ sn,
                              float* __restrict__ outk, float* __restrict__ outv,
                              unsigned short* __restrict__ kb, unsigned short* __restrict__ vtb,
                              int B, int L) {
  int task = blockIdx.x * 8 + (threadIdx.x >> 5);     // task = b*L + l
  int lane = threadIdx.x & 31;
  if (task >= B * L) return;
  int l = task % L;
  int b = task / L;
  const float* row = kf + (size_t)task * 128;
  float x0 = row[lane], x1 = row[lane + 32], x2 = row[lane + 64], x3 = row[lane + 96];
  float c0 = cs[(size_t)l * 64 + lane], c1 = cs[(size_t)l * 64 + lane + 32];
  float s0 = sn[(size_t)l * 64 + lane], s1 = sn[(size_t)l * 64 + lane + 32];
  float y0 = x0 * c0 + x2 * s0;
  float y1 = x1 * c1 + x3 * s1;
  float y2 = -x0 * s0 + x2 * c0;
  float y3 = -x1 * s1 + x3 * c1;
  float ss = y0 * y0 + y1 * y1 + y2 * y2 + y3 * y3;
#pragma unroll
  for (int off = 16; off; off >>= 1) ss += __shfl_xor(ss, off, 32);
  float r = rsqrtf(ss * (1.0f / 128.0f) + 1.1920929e-07f);
  float z0 = y0 * r, z1 = y1 * r, z2 = y2 * r, z3 = y3 * r;
  float* ok = outk + (size_t)task * 128;
  ok[lane] = z0; ok[lane + 32] = z1; ok[lane + 64] = z2; ok[lane + 96] = z3;
  unsigned short* kp = kb + (size_t)task * 128;
  kp[lane] = f2bf(z0); kp[lane + 32] = f2bf(z1);
  kp[lane + 64] = f2bf(z2); kp[lane + 96] = f2bf(z3);

  const float* vr = vf + (size_t)task * 128;
  float v0 = vr[lane], v1 = vr[lane + 32], v2 = vr[lane + 64], v3 = vr[lane + 96];
  float* ov = outv + (size_t)task * 128;
  ov[lane] = v0; ov[lane + 32] = v1; ov[lane + 64] = v2; ov[lane + 96] = v3;
  // V^T: (B,128,L)
  vtb[((size_t)b * 128 + lane) * L + l] = f2bf(v0);
  vtb[((size_t)b * 128 + lane + 32) * L + l] = f2bf(v1);
  vtb[((size_t)b * 128 + lane + 64) * L + l] = f2bf(v2);
  vtb[((size_t)b * 128 + lane + 96) * L + l] = f2bf(v3);
}

// ---------------- Flash attention (transposed-score formulation) ----------------
// Wave handles 16 queries of one (b,h); block = 8 waves = 128 queries.
// S^T = K_chunk(32x128) @ Q^T(128x16)  -> C layout: (key, query)
// O^T = V^T(128xkc) @ P^T(kc x16)      -> C layout: (d, query)
__global__ __launch_bounds__(256) void k_attn(
    const unsigned short* __restrict__ qb,   // (B,H,L,128)
    const unsigned short* __restrict__ kb,   // (B,L,128)
    const unsigned short* __restrict__ vtb,  // (B,128,L)
    unsigned short* __restrict__ aob,        // (B*L, 2048) bf16, head-major cols
    int B, int L, int H) {
  const int lane = threadIdx.x & 31;
  const int wave = threadIdx.x >> 5;
  const int hi = lane >> 4;
  const int lr = lane & 15;
  const int bh = blockIdx.y;
  const int b = bh / H;
  const int h = bh % H;
  const int q0 = blockIdx.x * 128 + wave * 16;

  const unsigned short* Q = qb + (size_t)bh * L * 128;
  const unsigned short* Km = kb + (size_t)b * L * 128;
  const unsigned short* Vt = vtb + (size_t)b * 128 * L;

  // Q^T B-fragments over d (held for the whole key loop)
  Frag qfr[4];
#pragma unroll
  for (int dj = 0; dj < 4; ++dj) qfr[dj] = load_b_frag(Q, 128, q0 + lr, 32 * dj, hi);

  v8f o[8];
#pragma unroll
  for (int f = 0; f < 8; ++f) o[f] = vzero();
  float mrow = -__builtin_inff();
  float lsum = 0.0f;
  const float scale = 0.08838834764831845f;   // 1/sqrt(128)
  const int query = q0 + lr;

  for (int kc = 0; kc < q0 + 16; kc += 32) {
    v8f sc0 = vzero(), sc1 = vzero();
#pragma unroll
    for (int dj = 0; dj < 4; ++dj) {
      Frag ka0 = load_a_frag(Km, 128, kc + lr, 32 * dj, hi);
      Frag ka1 = load_a_frag(Km, 128, kc + 16 + lr, 32 * dj, hi);
      sc0 = wmma_bf16(ka0, qfr[dj], sc0);
      sc1 = wmma_bf16(ka1, qfr[dj], sc1);
    }
    const bool needMask = (kc + 31 > q0);
    float e0[8], e1[8];
    float cmax = -__builtin_inff();
#pragma unroll
    for (int r = 0; r < 8; ++r) {
      float a = sc0[r] * scale;
      float c = sc1[r] * scale;
      if (needMask) {
        int key0 = kc + r + 8 * hi;
        if (key0 > query) a = -__builtin_inff();
        if (key0 + 16 > query) c = -__builtin_inff();
      }
      e0[r] = a; e1[r] = c;
      cmax = fmaxf(cmax, fmaxf(a, c));
    }
    cmax = fmaxf(cmax, __shfl_xor(cmax, 16, 32));   // full 32-key max per query
    float mnew = fmaxf(mrow, cmax);
    float alpha = __expf(mrow - mnew);
    float csum = 0.0f;
#pragma unroll
    for (int r = 0; r < 8; ++r) {
      e0[r] = __expf(e0[r] - mnew);
      e1[r] = __expf(e1[r] - mnew);
      csum += e0[r] + e1[r];
    }
    csum += __shfl_xor(csum, 16, 32);
    lsum = lsum * alpha + csum;
    mrow = mnew;
#pragma unroll
    for (int f = 0; f < 8; ++f) o[f] = o[f] * alpha;

    // Build P^T B-fragment (32 keys x 16 queries) from the S^T C layout.
    Frag pb;
#pragma unroll
    for (int j = 0; j < 4; ++j) {
      unsigned int p0 = pkbf(e0[2 * j], e0[2 * j + 1]);
      unsigned int p1 = pkbf(e1[2 * j], e1[2 * j + 1]);
      unsigned int x0 = __shfl_xor(p0, 16, 32);
      unsigned int x1 = __shfl_xor(p1, 16, 32);
      pb.u[j]     = hi ? x1 : p0;   // lanes<16: keys 0..7 ; lanes>=16: keys 16..23
      pb.u[j + 4] = hi ? p1 : x0;   // lanes<16: keys 8..15; lanes>=16: keys 24..31
    }
#pragma unroll
    for (int df = 0; df < 8; ++df) {
      Frag va = load_a_frag(Vt, L, 16 * df + lr, kc, hi);
      o[df] = wmma_bf16(va, pb, o[df]);
    }
  }

  float inv = 1.0f / lsum;
  size_t rowoff = ((size_t)(b * L) + q0 + lr) * 2048 + h * 128 + 8 * hi;
#pragma unroll
  for (int df = 0; df < 8; ++df) {
    uint4 st;
    st.x = pkbf(o[df][0] * inv, o[df][1] * inv);
    st.y = pkbf(o[df][2] * inv, o[df][3] * inv);
    st.z = pkbf(o[df][4] * inv, o[df][5] * inv);
    st.w = pkbf(o[df][6] * inv, o[df][7] * inv);
    *(uint4*)(aob + rowoff + 16 * df) = st;
  }
}

extern "C" void kernel_launch(void* const* d_in, const int* in_sizes, int n_in,
                              void* d_out, int out_size, void* d_ws, size_t ws_size,
                              hipStream_t stream) {
  const int B = 4, L = 2048, D = 2048, H = 16, HD = 128;
  const float* x  = (const float*)d_in[0];
  const float* cs = (const float*)d_in[1];
  const float* sn = (const float*)d_in[2];
  const float* Wq = (const float*)d_in[3];
  const float* Wk = (const float*)d_in[4];
  const float* Wv = (const float*)d_in[5];
  const float* Wo = (const float*)d_in[6];
  float* out  = (float*)d_out;
  float* outK = out + (size_t)B * L * D;
  float* outV = outK + (size_t)B * L * HD;

  char* p = (char*)d_ws;
  auto carve = [&](size_t bytes) { char* r = p; p += (bytes + 255) & ~(size_t)255; return r; };
  unsigned short* xb  = (unsigned short*)carve((size_t)B * L * D * 2);   // 33.5 MB
  unsigned short* wqb = (unsigned short*)carve((size_t)D * D * 2);       //  8.4 MB
  unsigned short* wkb = (unsigned short*)carve((size_t)HD * D * 2);
  unsigned short* wvb = (unsigned short*)carve((size_t)HD * D * 2);
  unsigned short* wob = (unsigned short*)carve((size_t)D * D * 2);       //  8.4 MB
  float* qf = (float*)carve((size_t)B * L * D * 4);                      // 67.1 MB
  float* kf = (float*)carve((size_t)B * L * HD * 4);
  float* vf = (float*)carve((size_t)B * L * HD * 4);
  unsigned short* qbb = (unsigned short*)carve((size_t)B * L * D * 2);   // 33.5 MB
  unsigned short* kbb = (unsigned short*)carve((size_t)B * L * HD * 2);
  unsigned short* vtb = (unsigned short*)carve((size_t)B * HD * L * 2);
  unsigned short* aob = (unsigned short*)qf;   // reuse dead q-f32 buffer

  auto cvt = [&](const float* in, unsigned short* o, size_t n) {
    k_cvt_bf16<<<(unsigned)((n + 255) / 256), 256, 0, stream>>>(in, o, (int)n);
  };
  cvt(x,  xb,  (size_t)B * L * D);
  cvt(Wq, wqb, (size_t)D * D);
  cvt(Wk, wkb, (size_t)HD * D);
  cvt(Wv, wvb, (size_t)HD * D);
  cvt(Wo, wob, (size_t)D * D);

  dim3 blk(256);
  k_gemm_bf16_nt<<<dim3(D / 128, (B * L) / 128), blk, 0, stream>>>(xb, wqb, qf, B * L, D, D);
  k_gemm_bf16_nt<<<dim3(HD / 128, (B * L) / 128), blk, 0, stream>>>(xb, wkb, kf, B * L, HD, D);
  k_gemm_bf16_nt<<<dim3(HD / 128, (B * L) / 128), blk, 0, stream>>>(xb, wvb, vf, B * L, HD, D);

  k_rope_rms_q<<<(B * L * H) / 8, blk, 0, stream>>>(qf, cs, sn, qbb, B, L, H);
  k_rope_rms_kv<<<(B * L) / 8, blk, 0, stream>>>(kf, vf, cs, sn, outK, outV, kbb, vtb, B, L);

  k_attn<<<dim3(L / 128, B * H), blk, 0, stream>>>(qbb, kbb, vtb, aob, B, L, H);

  k_gemm_bf16_nt<<<dim3(D / 128, (B * L) / 128), blk, 0, stream>>>(aob, wob, out, B * L, D, D);
}
